// DoubleNet_4526895530392
// MI455X (gfx1250) — matrix operand order repaired and checked by reading, not compiled, not run on hardware
//
#include <hip/hip_runtime.h>

#define N_ROWS 400000
#define D_IN   256
#define H_HID  512
#define BINS   10
#define SEGS   8192
#define BN_EPS 1e-5f
#define MTILE  128
#define SMEM_USHORTS (2 * MTILE * D_IN + 2 * 8 * 16 * 32)
#define SMEM_BYTES   (SMEM_USHORTS * 2)

typedef __attribute__((ext_vector_type(16))) __bf16        v16bf;
typedef __attribute__((ext_vector_type(8)))  float         v8f;
typedef __attribute__((ext_vector_type(8)))  unsigned int  v8u;

__device__ __forceinline__ unsigned short f2bf(float f) {
  unsigned int u = __builtin_bit_cast(unsigned int, f);
  u += 0x7fffu + ((u >> 16) & 1u);            // round-to-nearest-even
  return (unsigned short)(u >> 16);
}

// A-fragment (16-bit, 16x32): lane L (m = L%16, hi = L/16) holds K pairs
// {base, base+2, base+4, base+6, base+16, base+18, base+20, base+22}, base = k0 + hi*8.
__device__ __forceinline__ v8u load_afrag(const unsigned short* rowp, int base) {
  v8u r;
  r[0] = *(const unsigned int*)(rowp + base + 0);
  r[1] = *(const unsigned int*)(rowp + base + 2);
  r[2] = *(const unsigned int*)(rowp + base + 4);
  r[3] = *(const unsigned int*)(rowp + base + 6);
  r[4] = *(const unsigned int*)(rowp + base + 16);
  r[5] = *(const unsigned int*)(rowp + base + 18);
  r[6] = *(const unsigned int*)(rowp + base + 20);
  r[7] = *(const unsigned int*)(rowp + base + 22);
  return r;
}

// B-fragment (16-bit, 32x16) from K-contiguous transposed weights WT[n][k]:
// lane L holds column n = n0 + L%16, K halves kbase..kbase+7, kbase+16..kbase+23, kbase = k0 + (L/16)*8.
__device__ __forceinline__ v8u load_bfrag(const unsigned short* __restrict__ wt,
                                          int ldk, int n, int kbase) {
  const uint4* p = (const uint4*)(wt + (size_t)n * ldk + kbase);
  const uint4 x = p[0];
  const uint4 y = p[2];
  v8u r;
  r[0] = x.x; r[1] = x.y; r[2] = x.z; r[3] = x.w;
  r[4] = y.x; r[5] = y.y; r[6] = y.z; r[7] = y.w;
  return r;
}

__device__ __forceinline__ v8f wmma_bf16(v8u a, v8u b, v8f c) {
  return __builtin_amdgcn_wmma_f32_16x16x32_bf16(
      false, __builtin_bit_cast(v16bf, a),
      false, __builtin_bit_cast(v16bf, b),
      (short)0, c, false, false);
}

// ---------------------------------------------------------------------------
// Prep: bf16 transposed weights, BN affine fold, zero accumulators.
// ---------------------------------------------------------------------------
__global__ void prep_kernel(const float* __restrict__ W1q, const float* __restrict__ W1w,
                            const float* __restrict__ W2q, const float* __restrict__ W2w,
                            const float* __restrict__ gamma, const float* __restrict__ beta,
                            const float* __restrict__ rmean, const float* __restrict__ rvar,
                            unsigned short* __restrict__ WqT, unsigned short* __restrict__ WwT,
                            unsigned short* __restrict__ W2qT, unsigned short* __restrict__ W2wT,
                            float* __restrict__ affa, float* __restrict__ affb,
                            float* __restrict__ sums, float* __restrict__ lossp) {
  const int i = blockIdx.x * blockDim.x + threadIdx.x;
  if (i < H_HID * D_IN) {                 // W1 (D,H) -> WT[n][k], k contiguous
    const int n = i >> 8;
    const int k = i & (D_IN - 1);
    WqT[i] = f2bf(W1q[k * H_HID + n]);
    WwT[i] = f2bf(W1w[k * H_HID + n]);
  }
  if (i < 16 * H_HID) {                   // W2 padded to 16 output cols
    const int c = i >> 9;
    const int k = i & (H_HID - 1);
    W2qT[i] = (c < BINS) ? f2bf(W2q[k * BINS + c]) : (unsigned short)0;
    W2wT[i] = (c == 0) ? f2bf(W2w[k]) : (unsigned short)0;
  }
  if (i < D_IN) {                         // BN eval folded to x*a + b
    const float a = gamma[i] * rsqrtf(rvar[i] + BN_EPS);
    affa[i] = a;
    affb[i] = beta[i] - rmean[i] * a;
  }
  if (i < SEGS * BINS) sums[i] = 0.f;
  if (i == 0) *lossp = 0.f;
}

// ---------------------------------------------------------------------------
// Per-row segment id via binary search over boundaries.
// ---------------------------------------------------------------------------
__global__ void seg_ids_kernel(const int* __restrict__ sep, int* __restrict__ ids) {
  const int i = blockIdx.x * blockDim.x + threadIdx.x;
  if (i >= N_ROWS) return;
  int lo = 0, hi = SEGS;
  while (hi - lo > 1) {
    const int mid = (lo + hi) >> 1;
    if (sep[mid] <= i) lo = mid; else hi = mid;
  }
  ids[i] = lo;
}

// ---------------------------------------------------------------------------
// Fused dual-MLP + gating + segment accumulation. 8 waves x 16 rows per block.
// ---------------------------------------------------------------------------
__global__ __launch_bounds__(256) void fused_mlp_kernel(
    const float* __restrict__ data, const int* __restrict__ ids,
    const unsigned short* __restrict__ WqT, const unsigned short* __restrict__ WwT,
    const unsigned short* __restrict__ W2qT, const unsigned short* __restrict__ W2wT,
    const float* __restrict__ b1q, const float* __restrict__ b1w,
    const float* __restrict__ b2q, const float* __restrict__ b2w,
    const float* __restrict__ affa, const float* __restrict__ affb,
    float* __restrict__ sums) {
  extern __shared__ unsigned short smem[];
  unsigned short* lds_a  = smem;                      // [128][256] bf16 data
  unsigned short* lds_an = smem + MTILE * D_IN;       // [128][256] bf16 BN(data)
  unsigned short* hqbuf  = smem + 2 * MTILE * D_IN;   // [8][16][32] hidden (quality)
  unsigned short* hwbuf  = hqbuf + 8 * 16 * 32;       // [8][16][32] hidden (weight)

  const int tid  = threadIdx.x;
  const int wave = tid >> 5;
  const int lane = tid & 31;
  const int mloc = lane & 15;
  const int hi   = lane >> 4;
  const long row0 = (long)blockIdx.x * MTILE;

  // Stage tile: coalesced float4 loads, convert to bf16 (raw + BN-normalized).
  for (int e = tid * 4; e < MTILE * D_IN; e += 256 * 4) {
    const float4 v  = *(const float4*)(data + row0 * D_IN + e);
    const int    k  = e & (D_IN - 1);
    const float4 sa = *(const float4*)(affa + k);
    const float4 sb = *(const float4*)(affb + k);
    uint2 pa, pn;
    pa.x = (unsigned)f2bf(v.x) | ((unsigned)f2bf(v.y) << 16);
    pa.y = (unsigned)f2bf(v.z) | ((unsigned)f2bf(v.w) << 16);
    pn.x = (unsigned)f2bf(v.x * sa.x + sb.x) | ((unsigned)f2bf(v.y * sa.y + sb.y) << 16);
    pn.y = (unsigned)f2bf(v.z * sa.z + sb.z) | ((unsigned)f2bf(v.w * sa.w + sb.w) << 16);
    *(uint2*)(lds_a  + e) = pa;
    *(uint2*)(lds_an + e) = pn;
  }
  __syncthreads();

  const unsigned short* rowa  = lds_a  + (wave * 16 + mloc) * D_IN;
  const unsigned short* rowan = lds_an + (wave * 16 + mloc) * D_IN;
  unsigned short* myhq = hqbuf + wave * 16 * 32;
  unsigned short* myhw = hwbuf + wave * 16 * 32;

  v8f acc2q = {0.f, 0.f, 0.f, 0.f, 0.f, 0.f, 0.f, 0.f};
  v8f acc2w = {0.f, 0.f, 0.f, 0.f, 0.f, 0.f, 0.f, 0.f};

  for (int kk = 0; kk < 16; ++kk) {            // layer-2 K chunks (32 hidden each)
    for (int sub = 0; sub < 2; ++sub) {        // two 16-wide layer-1 N tiles
      const int n0   = kk * 32 + sub * 16;
      const int ncol = n0 + mloc;
      v8f accq = {0.f, 0.f, 0.f, 0.f, 0.f, 0.f, 0.f, 0.f};
      v8f accw = {0.f, 0.f, 0.f, 0.f, 0.f, 0.f, 0.f, 0.f};
      for (int kt = 0; kt < 8; ++kt) {         // K = 256 in steps of 32
        const int k0    = kt * 32;
        const int abase = k0 + hi * 8;
        const int kbase = k0 + hi * 8;
        v8u aq = load_afrag(rowa, abase);
        v8u bq = load_bfrag(WqT, D_IN, ncol, kbase);
        accq = wmma_bf16(aq, bq, accq);
        v8u aw = load_afrag(rowan, abase);
        v8u bw = load_bfrag(WwT, D_IN, ncol, kbase);
        accw = wmma_bf16(aw, bw, accw);
      }
      const float bq1 = b1q[ncol];
      const float bw1 = b1w[ncol];
#pragma unroll
      for (int v = 0; v < 8; ++v) {            // bias + ReLU, D-layout -> LDS tile
        float hq = accq[v] + bq1; hq = hq > 0.f ? hq : 0.f;
        float hw = accw[v] + bw1; hw = hw > 0.f ? hw : 0.f;
        const int r = v + 8 * hi;
        myhq[r * 32 + sub * 16 + mloc] = f2bf(hq);
        myhw[r * 32 + sub * 16 + mloc] = f2bf(hw);
      }
    }
    asm volatile("s_wait_dscnt 0" ::: "memory");  // LDS transpose round-trip
    v8u a2q  = load_afrag(myhq + mloc * 32, hi * 8);
    v8u b2qf = load_bfrag(W2qT, H_HID, mloc, kk * 32 + hi * 8);
    acc2q = wmma_bf16(a2q, b2qf, acc2q);
    v8u a2w  = load_afrag(myhw + mloc * 32, hi * 8);
    v8u b2wf = load_bfrag(W2wT, H_HID, mloc, kk * 32 + hi * 8);
    acc2w = wmma_bf16(a2w, b2wf, acc2w);
  }

  // Sigmoid gate lives in column 0 (lanes 0 / 16); broadcast per half-wave.
  const float b2wv = b2w[0];
  float wsc[8];
#pragma unroll
  for (int v = 0; v < 8; ++v) {
    const float x = acc2w[v] + b2wv;
    wsc[v] = 1.f / (1.f + expf(-x));
  }
#pragma unroll
  for (int v = 0; v < 8; ++v) wsc[v] = __shfl(wsc[v], hi << 4, 32);

  if (mloc < BINS) {
    const float bias2 = b2q[mloc];
#pragma unroll
    for (int v = 0; v < 8; ++v) {
      const long grow = row0 + wave * 16 + v + 8 * hi;
      const float q = (acc2q[v] + bias2) * wsc[v];
      atomicAdd(&sums[(long)ids[grow] * BINS + mloc], q);
    }
  }
}

// ---------------------------------------------------------------------------
// Segment mean -> softmax / log-softmax -> out_probs + NLL loss.
// ---------------------------------------------------------------------------
__global__ void finalize_kernel(const float* __restrict__ sums, const int* __restrict__ sep,
                                const int* __restrict__ tag, float* __restrict__ out) {
  const int s = blockIdx.x * blockDim.x + threadIdx.x;
  if (s >= SEGS) return;
  const float inv_cnt = 1.f / (float)(sep[s + 1] - sep[s]);
  float m[BINS];
  float mx = -3.4e38f;
#pragma unroll
  for (int c = 0; c < BINS; ++c) {
    m[c] = sums[s * BINS + c] * inv_cnt;
    mx = fmaxf(mx, m[c]);
  }
  float se = 0.f;
#pragma unroll
  for (int c = 0; c < BINS; ++c) se += expf(m[c] - mx);
  const float inv_se = 1.f / se;
#pragma unroll
  for (int c = 0; c < BINS; ++c) out[s * BINS + c] = expf(m[c] - mx) * inv_se;
  const int t = tag[s];
  const float logp = (m[t] - mx) - logf(se);
  atomicAdd(out + SEGS * BINS, -logp * (1.f / (float)SEGS));
}

// ---------------------------------------------------------------------------
extern "C" void kernel_launch(void* const* d_in, const int* in_sizes, int n_in,
                              void* d_out, int out_size, void* d_ws, size_t ws_size,
                              hipStream_t stream) {
  (void)in_sizes; (void)n_in; (void)out_size; (void)ws_size;
  const float* data  = (const float*)d_in[0];
  const int*   sep   = (const int*)d_in[1];
  const int*   tag   = (const int*)d_in[2];
  const float* W1q   = (const float*)d_in[3];
  const float* b1q   = (const float*)d_in[4];
  const float* W2q   = (const float*)d_in[5];
  const float* b2q   = (const float*)d_in[6];
  const float* gamma = (const float*)d_in[7];
  const float* beta  = (const float*)d_in[8];
  const float* rmean = (const float*)d_in[9];
  const float* rvar  = (const float*)d_in[10];
  const float* W1w   = (const float*)d_in[11];
  const float* b1w   = (const float*)d_in[12];
  const float* W2w   = (const float*)d_in[13];
  const float* b2w   = (const float*)d_in[14];
  float* out = (float*)d_out;

  char* ws = (char*)d_ws;
  size_t off = 0;
  float* sums = (float*)(ws + off);            off += (size_t)SEGS * BINS * 4;   // 320 KB
  int*   ids  = (int*)(ws + off);              off += (size_t)N_ROWS * 4;        // 1.6 MB
  unsigned short* WqT  = (unsigned short*)(ws + off); off += (size_t)H_HID * D_IN * 2;
  unsigned short* WwT  = (unsigned short*)(ws + off); off += (size_t)H_HID * D_IN * 2;
  unsigned short* W2qT = (unsigned short*)(ws + off); off += (size_t)16 * H_HID * 2;
  unsigned short* W2wT = (unsigned short*)(ws + off); off += (size_t)16 * H_HID * 2;
  float* affa = (float*)(ws + off);            off += 1024;
  float* affb = (float*)(ws + off);            off += 1024;

  (void)hipFuncSetAttribute(reinterpret_cast<const void*>(fused_mlp_kernel),
                            hipFuncAttributeMaxDynamicSharedMemorySize, SMEM_BYTES);

  prep_kernel<<<(H_HID * D_IN + 255) / 256, 256, 0, stream>>>(
      W1q, W1w, W2q, W2w, gamma, beta, rmean, rvar,
      WqT, WwT, W2qT, W2wT, affa, affb, sums, out + SEGS * BINS);
  seg_ids_kernel<<<(N_ROWS + 255) / 256, 256, 0, stream>>>(sep, ids);
  fused_mlp_kernel<<<N_ROWS / MTILE, 256, SMEM_BYTES, stream>>>(
      data, ids, WqT, WwT, W2qT, W2wT, b1q, b1w, b2q, b2w, affa, affb, sums);
  finalize_kernel<<<(SEGS + 255) / 256, 256, 0, stream>>>(sums, sep, tag, out);
}
